// GLANT_68865505624453
// MI455X (gfx1250) — matrix operand order repaired and checked by reading, not compiled, not run on hardware
//
#include <hip/hip_runtime.h>
#include <hip/hip_bf16.h>
#include <math.h>

typedef __attribute__((ext_vector_type(8)))  _Float16 v8h;
typedef __attribute__((ext_vector_type(16))) _Float16 v16h;
typedef __attribute__((ext_vector_type(8)))  float    v8f;

#define NNODES 50000
#define NEDGES 800000
#define DIN    128
#define HID    64
#define NOUT   40
#define SLOPE  0.2f
#define LN_EPS 1e-5f

// ---------------------------------------------------------------- utilities

__device__ __forceinline__ float warp_sum(float v) {
#pragma unroll
  for (int o = 16; o > 0; o >>= 1) v += __shfl_xor(v, o, 32);
  return v;
}

// monotonic float -> uint mapping for atomicMax on floats
__device__ __forceinline__ unsigned f2key(float f) {
  unsigned u = __float_as_uint(f);
  return (u & 0x80000000u) ? ~u : (u | 0x80000000u);
}
__device__ __forceinline__ float key2f(unsigned k) {
  unsigned u = (k & 0x80000000u) ? (k & 0x7FFFFFFFu) : ~k;
  return __uint_as_float(u);
}

// ---------------------------------------------------------------- L2 normalize rows -> f16

__global__ void l2norm_kernel(const float* __restrict__ x, _Float16* __restrict__ xn,
                              int n, int d) {
  int wave = (int)((blockIdx.x * blockDim.x + threadIdx.x) >> 5);
  int lane = threadIdx.x & 31;
  if (wave >= n) return;                       // wave-uniform
  const float* row = x + (size_t)wave * d;
  float ss = 0.f;
  for (int c = lane; c < d; c += 32) { float v = row[c]; ss += v * v; }
  ss = warp_sum(ss);
  float inv = 1.0f / fmaxf(sqrtf(ss), 1e-12f);
  _Float16* orow = xn + (size_t)wave * d;
  for (int c = lane; c < d; c += 32) orow[c] = (_Float16)(row[c] * inv);
}

// ---------------------------------------------------------------- weight prep
// B[K][N] fp32 row-major -> Bt[Npad][K] f16 (transposed, zero padded columns)

__global__ void conv_weight_kernel(const float* __restrict__ B, _Float16* __restrict__ Bt,
                                   int K, int N, int Npad) {
  int idx = (int)(blockIdx.x * blockDim.x + threadIdx.x);
  if (idx >= Npad * K) return;
  int n = idx / K;
  int k = idx - n * K;
  Bt[idx] = (n < N) ? (_Float16)B[(size_t)k * N + n] : (_Float16)0.0f;
}

// ---------------------------------------------------------------- WMMA GEMM
// C[M,Nreal] = A[M,K](f16) * B(f16, pre-transposed Bt[Npad][K]) + bias.
// One wave per 16-row stripe: TN 16x16 tiles, A fragment reused across tiles.
// A/B fragments in contiguous k order (k = 16*half + j); C/D layout per CDNA5 ISA §7.12.2 (wave32).

template <int TN>
__global__ void wmma_gemm_kernel(const _Float16* __restrict__ A, int lda,
                                 const _Float16* __restrict__ Bt,
                                 const float* __restrict__ bias,
                                 float* __restrict__ C, int ldc,
                                 int M, int Nreal, int K) {
  int wavesPerBlock = blockDim.x >> 5;
  int tm = (int)blockIdx.x * wavesPerBlock + (int)(threadIdx.x >> 5);
  if (tm >= (M >> 4)) return;                  // wave-uniform
  int lane = threadIdx.x & 31;
  int hf  = lane >> 4;                         // lane half
  int l16 = lane & 15;
  int row = tm * 16 + l16;

  const _Float16* Arow = A + (size_t)row * lda;
  v8f acc[TN];
#pragma unroll
  for (int t = 0; t < TN; ++t) acc[t] = (v8f){};

  for (int k0 = 0; k0 < K; k0 += 32) {
    // A 16x32 fragment: K = k0 + hf*16 + j, contiguous (matches B)
    v8h alo = *(const v8h*)(Arow + k0 + hf * 16);
    v8h ahi = *(const v8h*)(Arow + k0 + hf * 16 + 8);
    v16h a = __builtin_shufflevector(alo, ahi, 0, 1, 2, 3, 4, 5, 6, 7,
                                     8, 9, 10, 11, 12, 13, 14, 15);
#pragma unroll
    for (int t = 0; t < TN; ++t) {
      // B 32x16 fragment: lane holds column t*16+l16; K = k0 + hf*16 + j, contiguous in Bt
      int col = t * 16 + l16;
      v16h b = *(const v16h*)(Bt + (size_t)col * K + k0 + hf * 16);
      acc[t] = __builtin_amdgcn_wmma_f32_16x16x32_f16(
          false, a, false, b, (short)0, acc[t], false, false);
    }
  }
#pragma unroll
  for (int t = 0; t < TN; ++t) {
    int col = t * 16 + l16;
    if (col < Nreal) {
      float bv = bias[col];
#pragma unroll
      for (int r = 0; r < 8; ++r) {
        int m = r + hf * 8;                    // C/D layout: VGPR r -> M=r (+8 for high half)
        C[(size_t)(tm * 16 + m) * ldc + col] = acc[t][r] + bv;
      }
    }
  }
}

// ---------------------------------------------------------------- GATv2 edge phases
// edges layout: ei[0..E) = src, ei[E..2E) = dst; e >= E encodes self loop (e-E, e-E)

__global__ void edge_logits_kernel(const float* __restrict__ xl, const float* __restrict__ xr,
                                   const float* __restrict__ att, int C,
                                   const int* __restrict__ ei, int E, int nself,
                                   float* __restrict__ logits, unsigned* __restrict__ mkey) {
  int e = (int)((blockIdx.x * blockDim.x + threadIdx.x) >> 5);
  int lane = threadIdx.x & 31;
  if (e >= E + nself) return;                  // wave-uniform
  int s, d;
  if (e < E) { s = ei[e]; d = ei[E + e]; } else { s = d = e - E; }
  float p = 0.f;
  int c = lane * 2;
  if (c < C) {
    float v0 = xl[(size_t)s * C + c]     + xr[(size_t)d * C + c];
    float v1 = xl[(size_t)s * C + c + 1] + xr[(size_t)d * C + c + 1];
    v0 = (v0 >= 0.f) ? v0 : SLOPE * v0;
    v1 = (v1 >= 0.f) ? v1 : SLOPE * v1;
    p = att[c] * v0 + att[c + 1] * v1;
  }
  p = warp_sum(p);
  if (lane == 0) {
    logits[e] = p;
    atomicMax(&mkey[d], f2key(p));
  }
}

__global__ void edge_expsum_kernel(float* __restrict__ logw, const int* __restrict__ ei,
                                   int E, int nself,
                                   const unsigned* __restrict__ mkey,
                                   float* __restrict__ denom) {
  int e = (int)(blockIdx.x * blockDim.x + threadIdx.x);
  if (e >= E + nself) return;
  int d = (e < E) ? ei[E + e] : (e - E);
  unsigned k = mkey[d];
  float m = (k == 0u) ? 0.0f : key2f(k);       // matches where(isneginf(m), 0, m)
  float w = expf(logw[e] - m);
  logw[e] = w;                                 // overwrite logits with weights
  atomicAdd(&denom[d], w);
}

__global__ void edge_agg_kernel(const float* __restrict__ xl, const float* __restrict__ logw,
                                const float* __restrict__ denom, const int* __restrict__ ei,
                                int E, int nself, int C, float* __restrict__ out) {
  int e = (int)((blockIdx.x * blockDim.x + threadIdx.x) >> 5);
  int lane = threadIdx.x & 31;
  if (e >= E + nself) return;                  // wave-uniform
  int s, d;
  if (e < E) { s = ei[e]; d = ei[E + e]; } else { s = d = e - E; }
  float alpha = logw[e] / denom[d];
  int c = lane * 2;
  if (c < C) {
    atomicAdd(&out[(size_t)d * C + c],     alpha * xl[(size_t)s * C + c]);
    atomicAdd(&out[(size_t)d * C + c + 1], alpha * xl[(size_t)s * C + c + 1]);
  }
}

// ---------------------------------------------------------------- layer-0 epilogue
// z = (aggA+bias0) + sigmoid(theta)*(aggB+bias1); LayerNorm; ReLU; + hproj -> x1 (f16)

__global__ void combine0_kernel(const float* __restrict__ aggA, const float* __restrict__ aggB,
                                const float* __restrict__ bias0, const float* __restrict__ bias1,
                                const float* __restrict__ theta,
                                const float* __restrict__ gamma, const float* __restrict__ beta,
                                const float* __restrict__ hproj,
                                _Float16* __restrict__ x1, int n) {
  int node = (int)((blockIdx.x * blockDim.x + threadIdx.x) >> 5);
  int lane = threadIdx.x & 31;
  if (node >= n) return;                       // wave-uniform
  float sig = 1.0f / (1.0f + expf(-theta[0]));
  size_t base = (size_t)node * HID;
  int c0 = lane * 2, c1 = c0 + 1;              // HID==64 -> exactly 2 per lane
  float z0 = aggA[base + c0] + bias0[c0] + sig * (aggB[base + c0] + bias1[c0]);
  float z1 = aggA[base + c1] + bias0[c1] + sig * (aggB[base + c1] + bias1[c1]);
  float mu = warp_sum(z0 + z1) * (1.0f / HID);
  float d0 = z0 - mu, d1 = z1 - mu;
  float var = warp_sum(d0 * d0 + d1 * d1) * (1.0f / HID);
  float inv = 1.0f / sqrtf(var + LN_EPS);
  z0 = fmaxf(d0 * inv * gamma[c0] + beta[c0], 0.f);
  z1 = fmaxf(d1 * inv * gamma[c1] + beta[c1], 0.f);
  x1[base + c0] = (_Float16)(z0 + hproj[base + c0]);
  x1[base + c1] = (_Float16)(z1 + hproj[base + c1]);
}

// ---------------------------------------------------------------- layer-1 epilogue

__global__ void combine1_kernel(const float* __restrict__ aggA, const float* __restrict__ aggB,
                                const float* __restrict__ bias0, const float* __restrict__ bias1,
                                const float* __restrict__ theta,
                                float* __restrict__ out, int n, int C) {
  int i = (int)(blockIdx.x * blockDim.x + threadIdx.x);
  if (i >= n * C) return;
  int c = i % C;
  float sig = 1.0f / (1.0f + expf(-theta[0]));
  out[i] = aggA[i] + bias0[c] + sig * (aggB[i] + bias1[c]);
}

// ---------------------------------------------------------------- host launch

extern "C" void kernel_launch(void* const* d_in, const int* in_sizes, int n_in,
                              void* d_out, int out_size, void* d_ws, size_t ws_size,
                              hipStream_t stream) {
  (void)in_sizes; (void)n_in; (void)out_size; (void)ws_size;

  const float* x        = (const float*)d_in[0];
  const int*   ei0      = (const int*)d_in[1];
  const int*   ei1      = (const int*)d_in[2];
  const float* l0_Wl    = (const float*)d_in[3];
  const float* l0_bl    = (const float*)d_in[4];
  const float* l0_Wr0   = (const float*)d_in[5];
  const float* l0_br0   = (const float*)d_in[6];
  const float* l0_att0  = (const float*)d_in[7];
  const float* l0_bias0 = (const float*)d_in[8];
  const float* l0_Wr1   = (const float*)d_in[9];
  const float* l0_br1   = (const float*)d_in[10];
  const float* l0_att1  = (const float*)d_in[11];
  const float* l0_bias1 = (const float*)d_in[12];
  const float* l0_theta = (const float*)d_in[13];
  const float* l1_Wl    = (const float*)d_in[14];
  const float* l1_bl    = (const float*)d_in[15];
  const float* l1_Wr0   = (const float*)d_in[16];
  const float* l1_br0   = (const float*)d_in[17];
  const float* l1_att0  = (const float*)d_in[18];
  const float* l1_bias0 = (const float*)d_in[19];
  const float* l1_Wr1   = (const float*)d_in[20];
  const float* l1_br1   = (const float*)d_in[21];
  const float* l1_att1  = (const float*)d_in[22];
  const float* l1_bias1 = (const float*)d_in[23];
  const float* l1_theta = (const float*)d_in[24];
  const float* ln_gamma = (const float*)d_in[25];
  const float* ln_beta  = (const float*)d_in[26];
  const float* proj_W   = (const float*)d_in[27];
  const float* proj_b   = (const float*)d_in[28];
  float* out = (float*)d_out;

  // workspace layout (256B aligned slices)
  char* ws = (char*)d_ws;
  size_t off = 0;
  auto alloc = [&](size_t bytes) -> void* {
    void* p = (void*)(ws + off);
    off += (bytes + 255) & ~(size_t)255;
    return p;
  };
  _Float16* xn_h  = (_Float16*)alloc((size_t)NNODES * DIN * 2);
  _Float16* x1_h  = (_Float16*)alloc((size_t)NNODES * HID * 2);
  float* xl0   = (float*)alloc((size_t)NNODES * HID * 4);
  float* xr00  = (float*)alloc((size_t)NNODES * HID * 4);
  float* xr01  = (float*)alloc((size_t)NNODES * HID * 4);
  float* hproj = (float*)alloc((size_t)NNODES * HID * 4);
  float* aggA  = (float*)alloc((size_t)NNODES * HID * 4);
  float* aggB  = (float*)alloc((size_t)NNODES * HID * 4);
  float* xl1   = (float*)alloc((size_t)NNODES * NOUT * 4);
  float* xr10  = (float*)alloc((size_t)NNODES * NOUT * 4);
  float* xr11  = (float*)alloc((size_t)NNODES * NOUT * 4);
  float* logw  = (float*)alloc((size_t)(NEDGES + NNODES) * 4);
  unsigned* mkey = (unsigned*)alloc((size_t)NNODES * 4);
  float* denom = (float*)alloc((size_t)NNODES * 4);
  // padded/transposed f16 weights
  const int NP0 = 64, NP1 = 48;                       // padded N for layer0 / layer1
  _Float16* BtWl0 = (_Float16*)alloc((size_t)NP0 * DIN * 2);
  _Float16* BtWr00 = (_Float16*)alloc((size_t)NP0 * DIN * 2);
  _Float16* BtWr01 = (_Float16*)alloc((size_t)NP0 * DIN * 2);
  _Float16* BtProj = (_Float16*)alloc((size_t)NP0 * DIN * 2);
  _Float16* BtWl1 = (_Float16*)alloc((size_t)NP1 * HID * 2);
  _Float16* BtWr10 = (_Float16*)alloc((size_t)NP1 * HID * 2);
  _Float16* BtWr11 = (_Float16*)alloc((size_t)NP1 * HID * 2);

  auto convw = [&](const float* B, _Float16* Bt, int K, int N, int Npad) {
    int n = Npad * K;
    conv_weight_kernel<<<(n + 255) / 256, 256, 0, stream>>>(B, Bt, K, N, Npad);
  };

  auto gat = [&](const float* xl, const float* xr, const float* att, int C,
                 const int* ei, int nself, float* agg) {
    hipMemsetAsync(mkey, 0, (size_t)NNODES * 4, stream);
    hipMemsetAsync(denom, 0, (size_t)NNODES * 4, stream);
    hipMemsetAsync(agg, 0, (size_t)NNODES * C * 4, stream);
    int total = NEDGES + nself;
    int wpb = 8;  // 256 threads = 8 waves/block (one edge per wave)
    edge_logits_kernel<<<(total + wpb - 1) / wpb, 256, 0, stream>>>(
        xl, xr, att, C, ei, NEDGES, nself, logw, mkey);
    edge_expsum_kernel<<<(total + 255) / 256, 256, 0, stream>>>(
        logw, ei, NEDGES, nself, mkey, denom);
    edge_agg_kernel<<<(total + wpb - 1) / wpb, 256, 0, stream>>>(
        xl, logw, denom, ei, NEDGES, nself, C, agg);
  };

  const int MT = NNODES / 16;                  // 3125 row tiles (exact)
  const int GW = 8;                            // waves per 256-thread block
  const int gemmGrid = (MT + GW - 1) / GW;

  // 0) weight conversions (tiny)
  convw(l0_Wl,  BtWl0,  DIN, HID, NP0);
  convw(l0_Wr0, BtWr00, DIN, HID, NP0);
  convw(l0_Wr1, BtWr01, DIN, HID, NP0);
  convw(proj_W, BtProj, DIN, HID, NP0);
  convw(l1_Wl,  BtWl1,  HID, NOUT, NP1);
  convw(l1_Wr0, BtWr10, HID, NOUT, NP1);
  convw(l1_Wr1, BtWr11, HID, NOUT, NP1);

  // 1) x <- L2-normalize rows (writes f16)
  l2norm_kernel<<<(NNODES + 7) / 8, 256, 0, stream>>>(x, xn_h, NNODES, DIN);

  // 2) layer-0 GEMMs (K=128, N=64) via WMMA
  wmma_gemm_kernel<4><<<gemmGrid, 256, 0, stream>>>(xn_h, DIN, BtWl0,  l0_bl,  xl0,   HID, NNODES, HID, DIN);
  wmma_gemm_kernel<4><<<gemmGrid, 256, 0, stream>>>(xn_h, DIN, BtWr00, l0_br0, xr00,  HID, NNODES, HID, DIN);
  wmma_gemm_kernel<4><<<gemmGrid, 256, 0, stream>>>(xn_h, DIN, BtWr01, l0_br1, xr01,  HID, NNODES, HID, DIN);
  wmma_gemm_kernel<4><<<gemmGrid, 256, 0, stream>>>(xn_h, DIN, BtProj, proj_b, hproj, HID, NNODES, HID, DIN);

  // 3) layer-0 attention hops (hop0 has self loops)
  gat(xl0, xr00, l0_att0, HID, ei0, NNODES, aggA);
  gat(xl0, xr01, l0_att1, HID, ei1, 0,      aggB);

  // 4) combine + LayerNorm + ReLU + residual projection -> x1 (f16)
  combine0_kernel<<<(NNODES + 7) / 8, 256, 0, stream>>>(
      aggA, aggB, l0_bias0, l0_bias1, l0_theta, ln_gamma, ln_beta, hproj, x1_h, NNODES);

  // 5) layer-1 GEMMs (K=64, N=40) via WMMA
  wmma_gemm_kernel<3><<<gemmGrid, 256, 0, stream>>>(x1_h, HID, BtWl1,  l1_bl,  xl1,  NOUT, NNODES, NOUT, HID);
  wmma_gemm_kernel<3><<<gemmGrid, 256, 0, stream>>>(x1_h, HID, BtWr10, l1_br0, xr10, NOUT, NNODES, NOUT, HID);
  wmma_gemm_kernel<3><<<gemmGrid, 256, 0, stream>>>(x1_h, HID, BtWr11, l1_br1, xr11, NOUT, NNODES, NOUT, HID);

  // 6) layer-1 attention hops (reuse agg buffers, C=40)
  gat(xl1, xr10, l1_att0, NOUT, ei0, NNODES, aggA);
  gat(xl1, xr11, l1_att1, NOUT, ei1, 0,      aggB);

  // 7) final combine -> d_out
  combine1_kernel<<<((NNODES * NOUT) + 255) / 256, 256, 0, stream>>>(
      aggA, aggB, l1_bias0, l1_bias1, l1_theta, out, NNODES, NOUT);
}
